// GCN_59090160058839
// MI455X (gfx1250) — matrix operand order repaired and checked by reading, not compiled, not run on hardware
//
#include <hip/hip_runtime.h>
#include <hip/hip_bf16.h>

// ---------------------------------------------------------------------------
// GCN on gfx1250 (MI455X): bf16 WMMA GEMMs (LDS-staged A) + f32-atomic scatter.
// Inputs (setup_inputs order):
//  0 h[N,128] f32, 1 W1[128,128] f32, 2 b1[128] f32, 3 p[128] f32,
//  4 W2[128,128] f32, 5 b2[128] f32, 6 Wc[128,16] f32, 7 bc[16] f32,
//  8 src[E] i32, 9 dst[E] i32, 10 graph_ids[N] i32.  Output: [100,16] f32.
// ---------------------------------------------------------------------------

typedef __attribute__((ext_vector_type(16))) __bf16 v16bf;
typedef __attribute__((ext_vector_type(8)))  __bf16 v8bf;
typedef __attribute__((ext_vector_type(8)))  float  v8f;

#define DI        128
#define NGRAPH    100
#define NCLASS    16

// ---------------------------------------------------------------------------
// Degree accumulation (f32 atomics; degrees are small integers, exact in f32)
// ---------------------------------------------------------------------------
__global__ void gcn_degrees(const int* __restrict__ src, const int* __restrict__ dst,
                            float* __restrict__ degO, float* __restrict__ degI, int ne) {
    int e = blockIdx.x * blockDim.x + threadIdx.x;
    if (e >= ne) return;
    atomicAdd(&degO[src[e]], 1.0f);
    atomicAdd(&degI[dst[e]], 1.0f);
}

// deg -> (deg>0 ? rsqrt(max(deg,1)) : 0), in place
__global__ void gcn_to_norm(float* __restrict__ deg, int n) {
    int i = blockIdx.x * blockDim.x + threadIdx.x;
    if (i >= n) return;
    float d = deg[i];
    deg[i] = (d > 0.0f) ? rsqrtf(fmaxf(d, 1.0f)) : 0.0f;
}

// ---------------------------------------------------------------------------
// Pack a 128x128 f32 weight into a contiguous-k B-operand per-lane layout for
// v_wmma_f32_16x16x32_bf16 (same k order as the A loads):  Wp[ct][ks][lane][0..15], where
// lane (0..31) holds column n = ct*16 + lane%16, K = ks*32 + (lane/16)*16 + i.
// ---------------------------------------------------------------------------
__global__ void gcn_pack_w(const float* __restrict__ W, __bf16* __restrict__ Wp) {
    int t = blockIdx.x * blockDim.x + threadIdx.x;   // 0..1023
    if (t >= 8 * 4 * 32) return;
    int lane = t & 31;
    int ks   = (t >> 5) & 3;
    int ct   = t >> 7;
    int n  = ct * 16 + (lane & 15);
    int k0 = ks * 32 + ((lane >> 4) << 4);
    __bf16* dstp = Wp + (size_t)t * 16;
#pragma unroll
    for (int i = 0; i < 16; ++i)
        dstp[i] = (__bf16)W[(size_t)(k0 + i) * DI + n];
}

// ---------------------------------------------------------------------------
// Y[row, :] = (X[row, :] @ W) * rowscale[row]
// Block = 256 threads = 8 waves covering 16 rows x 128 cols.
// A stripe (16x128) is staged once per block into LDS as bf16 (4 KB); each
// wave then builds its A fragment with two 16B ds loads per K-step and runs
// 4x v_wmma_f32_16x16x32_bf16 (K = 128).
// ---------------------------------------------------------------------------
__global__ __launch_bounds__(256)
void gcn_gemm128(const float* __restrict__ X, const __bf16* __restrict__ Wp,
                 const float* __restrict__ rowscale, float* __restrict__ Y, int nrows) {
    __shared__ __bf16 As[16 * DI];                      // 4 KB

    const int tid     = threadIdx.x;
    const int rowBase = blockIdx.x * 16;

    // ---- stage A: 16 rows x 128 f32 -> bf16 LDS (one pass, one cvt/elem) ----
    {
        const int r  = tid >> 4;                        // 0..15
        const int c0 = (tid & 15) << 3;                 // 0,8,...,120
        const int gr = min(rowBase + r, nrows - 1);     // clamp: dup loads OK
        const float4* gp = (const float4*)(X + (size_t)gr * DI + c0);
        const float4 q0 = gp[0], q1 = gp[1];
        __bf16* lp = As + r * DI + c0;
        lp[0] = (__bf16)q0.x; lp[1] = (__bf16)q0.y;
        lp[2] = (__bf16)q0.z; lp[3] = (__bf16)q0.w;
        lp[4] = (__bf16)q1.x; lp[5] = (__bf16)q1.y;
        lp[6] = (__bf16)q1.z; lp[7] = (__bf16)q1.w;
    }
    __syncthreads();

    const int lane = tid & 31;
    const int wave = tid >> 5;                          // column tile 0..7
    const int m    = lane & 15;                         // row in tile / col in tile
    const int half = lane >> 4;                         // 0 or 1

    v8f acc = {};
#pragma unroll
    for (int ks = 0; ks < 4; ++ks) {
        const int k0 = ks * 32;
        // A fragment (16-bit A 16x32): lanes<16 hold K k0+{0..15},
        // lanes>=16 hold K k0+{16..31}; row M = lane%16.
        const int kA0 = k0 + half * 16;
        const int kA1 = kA0 + 8;
        const v8bf a0 = *(const v8bf*)(As + m * DI + kA0);   // 16B ds load
        const v8bf a1 = *(const v8bf*)(As + m * DI + kA1);   // 16B ds load
        const v16bf a = __builtin_shufflevector(a0, a1,
            0, 1, 2, 3, 4, 5, 6, 7, 8, 9, 10, 11, 12, 13, 14, 15);
        // B fragment: pre-packed, one aligned 32B vector per lane per K-step.
        const v16bf b = *(const v16bf*)(Wp + ((((size_t)wave * 4 + ks) * 32 + lane) * 16));
        acc = __builtin_amdgcn_wmma_f32_16x16x32_bf16(
                  false, a, false, b, (short)0, acc, false, false);
    }

    // D layout: VGPR j -> row rowBase + j + 8*half, col = 16*wave + lane%16.
    const int colBase = wave * 16;
    if (rowBase + 16 <= nrows) {                        // full tile: no guards
#pragma unroll
        for (int j = 0; j < 8; ++j) {
            const int r = rowBase + j + 8 * half;
            Y[(size_t)r * DI + colBase + m] = acc[j] * rowscale[r];
        }
    } else {
#pragma unroll
        for (int j = 0; j < 8; ++j) {
            const int r = rowBase + j + 8 * half;
            if (r < nrows)
                Y[(size_t)r * DI + colBase + m] = acc[j] * rowscale[r];
        }
    }
}

// ---------------------------------------------------------------------------
// Edge scatter: agg[dst[e], :] += msg[src[e], :].  One wave32 per edge,
// each lane moves a float4 (32 lanes x 16B = 512B row).
// ---------------------------------------------------------------------------
__global__ __launch_bounds__(256)
void gcn_scatter(const int* __restrict__ src, const int* __restrict__ dst,
                 const float* __restrict__ msg, float* __restrict__ agg, int ne) {
    const int e    = (blockIdx.x * blockDim.x + threadIdx.x) >> 5;
    const int lane = threadIdx.x & 31;
    if (e >= ne) return;
    const int s = src[e];
    const int d = dst[e];
    const float4 v = ((const float4*)(msg + (size_t)s * DI))[lane];
    float* a = agg + (size_t)d * DI + lane * 4;
    atomicAdd(a + 0, v.x);
    atomicAdd(a + 1, v.y);
    atomicAdd(a + 2, v.z);
    atomicAdd(a + 3, v.w);
}

// x = relu(agg * norm_dst + b1) * clamp(p, 0, 1)
__global__ void gcn_post1(const float* __restrict__ agg, const float* __restrict__ normD,
                          const float* __restrict__ b1, const float* __restrict__ p,
                          float* __restrict__ x, int nnodes) {
    const long i = (long)blockIdx.x * blockDim.x + threadIdx.x;
    const long n = (long)nnodes * DI;
    if (i >= n) return;
    const int f = (int)(i & (DI - 1));
    const long node = i >> 7;
    float v = fmaf(agg[i], normD[node], b1[f]);
    v = fmaxf(v, 0.0f);
    const float pc = fminf(fmaxf(p[f], 0.0f), 1.0f);
    x[i] = v * pc;
}

// Fused layer-2 epilogue + graph pooling:
// gsum[g, :] += agg2[node, :] * norm_dst[node] + b2 ; gcnt[g] += 1
__global__ __launch_bounds__(256)
void gcn_pool(const float* __restrict__ agg2, const float* __restrict__ normD,
              const float* __restrict__ b2, const int* __restrict__ gid,
              float* __restrict__ gsum, float* __restrict__ gcnt, int nnodes) {
    const int node = (blockIdx.x * blockDim.x + threadIdx.x) >> 5;
    const int lane = threadIdx.x & 31;
    if (node >= nnodes) return;
    const int g = gid[node];
    const float ns = normD[node];
    const float4 v = ((const float4*)(agg2 + (size_t)node * DI))[lane];
    const float4 b = ((const float4*)b2)[lane];
    float* gs = gsum + (size_t)g * DI + lane * 4;
    atomicAdd(gs + 0, fmaf(v.x, ns, b.x));
    atomicAdd(gs + 1, fmaf(v.y, ns, b.y));
    atomicAdd(gs + 2, fmaf(v.z, ns, b.z));
    atomicAdd(gs + 3, fmaf(v.w, ns, b.w));
    if (lane == 0) atomicAdd(&gcnt[g], 1.0f);
}

// out[g, c] = (gsum[g, :] / max(cnt, 1)) . Wc[:, c] + bc[c]
__global__ void gcn_classify(const float* __restrict__ gsum, const float* __restrict__ gcnt,
                             const float* __restrict__ Wc, const float* __restrict__ bc,
                             float* __restrict__ out) {
    const int t = blockIdx.x * blockDim.x + threadIdx.x;
    if (t >= NGRAPH * NCLASS) return;
    const int g = t >> 4, c = t & 15;
    const float inv = 1.0f / fmaxf(gcnt[g], 1.0f);
    float acc = 0.0f;
#pragma unroll 4
    for (int k = 0; k < DI; ++k)
        acc = fmaf(gsum[(size_t)g * DI + k], Wc[(size_t)k * NCLASS + c], acc);
    out[t] = fmaf(acc, inv, bc[c]);
}

// ---------------------------------------------------------------------------
extern "C" void kernel_launch(void* const* d_in, const int* in_sizes, int n_in,
                              void* d_out, int out_size, void* d_ws, size_t ws_size,
                              hipStream_t stream) {
    const float* h   = (const float*)d_in[0];
    const float* W1  = (const float*)d_in[1];
    const float* b1  = (const float*)d_in[2];
    const float* p   = (const float*)d_in[3];
    const float* W2  = (const float*)d_in[4];
    const float* b2  = (const float*)d_in[5];
    const float* Wc  = (const float*)d_in[6];
    const float* bc  = (const float*)d_in[7];
    const int*   src = (const int*)d_in[8];
    const int*   dst = (const int*)d_in[9];
    const int*   gid = (const int*)d_in[10];
    float* out = (float*)d_out;

    const int nNodes = in_sizes[0] / DI;
    const int nEdges = in_sizes[8];

    // Workspace layout (all offsets 32B-aligned)
    const size_t NB = (size_t)nNodes * DI * sizeof(float);
    char* ws = (char*)d_ws;
    float*  buf0 = (float*)ws;                                   // NB
    float*  buf1 = (float*)(ws + NB);                            // NB
    float*  degO = (float*)(ws + 2 * NB);                        // nNodes
    float*  degI = degO + nNodes;                                // nNodes
    __bf16* Wp1  = (__bf16*)(degI + nNodes);                     // 8*4*32*16 bf16
    __bf16* Wp2  = Wp1 + 8 * 4 * 32 * 16;
    float*  gsum = (float*)(Wp2 + 8 * 4 * 32 * 16);              // 100*128
    float*  gcnt = gsum + NGRAPH * DI;                           // 100

    // --- degrees -> symmetric norms ---
    hipMemsetAsync(degO, 0, (size_t)nNodes * sizeof(float), stream);
    hipMemsetAsync(degI, 0, (size_t)nNodes * sizeof(float), stream);
    gcn_degrees<<<(nEdges + 255) / 256, 256, 0, stream>>>(src, dst, degO, degI, nEdges);
    gcn_to_norm<<<(nNodes + 255) / 256, 256, 0, stream>>>(degO, nNodes);  // norm_src
    gcn_to_norm<<<(nNodes + 255) / 256, 256, 0, stream>>>(degI, nNodes);  // norm_dst

    // --- pack weights to WMMA B layout (bf16) ---
    gcn_pack_w<<<4, 256, 0, stream>>>(W1, Wp1);
    gcn_pack_w<<<4, 256, 0, stream>>>(W2, Wp2);

    const int gemmBlocks    = (nNodes + 15) / 16;
    const int scatterBlocks = (int)(((size_t)nEdges * 32 + 255) / 256);
    const long nFeat        = (long)nNodes * DI;

    // --- layer 1: buf0 = (h @ W1) * norm_src ; buf1 = scatter ; buf0 = act ---
    gcn_gemm128<<<gemmBlocks, 256, 0, stream>>>(h, Wp1, degO, buf0, nNodes);
    hipMemsetAsync(buf1, 0, NB, stream);
    gcn_scatter<<<scatterBlocks, 256, 0, stream>>>(src, dst, buf0, buf1, nEdges);
    gcn_post1<<<(int)((nFeat + 255) / 256), 256, 0, stream>>>(buf1, degI, b1, p, buf0, nNodes);

    // --- layer 2: buf1 = (x @ W2) * norm_src ; buf0 = scatter ---
    gcn_gemm128<<<gemmBlocks, 256, 0, stream>>>(buf0, Wp2, degO, buf1, nNodes);
    hipMemsetAsync(buf0, 0, NB, stream);
    gcn_scatter<<<scatterBlocks, 256, 0, stream>>>(src, dst, buf1, buf0, nEdges);

    // --- pooling (fused +b2, *norm_dst) and classifier ---
    hipMemsetAsync(gsum, 0, (size_t)NGRAPH * DI * sizeof(float), stream);
    hipMemsetAsync(gcnt, 0, (size_t)NGRAPH * sizeof(float), stream);
    gcn_pool<<<(int)(((size_t)nNodes * 32 + 255) / 256), 256, 0, stream>>>(
        buf0, degI, b2, gid, gsum, gcnt, nNodes);
    gcn_classify<<<(NGRAPH * NCLASS + 255) / 256, 256, 0, stream>>>(gsum, gcnt, Wc, bc, out);
}
